// MultiHeadAttention_10033043603552
// MI455X (gfx1250) — matrix operand order repaired and checked by reading, not compiled, not run on hardware
//
#include <hip/hip_runtime.h>

typedef _Float16 v16h __attribute__((ext_vector_type(16)));
typedef float    v8f  __attribute__((ext_vector_type(8)));

#define D_MODEL 1024
#define N_HEADS 16
#define D_K     64
#define BATCH   2
#define SEQ     2048
#define BS      (BATCH * SEQ)      // 4096
#define BH      (BATCH * N_HEADS)  // 32

// pack two fp32 into one dword of two f16 (lowers to v_cvt_pk_f16_f32)
__device__ __forceinline__ unsigned pk2(float lo, float hi) {
    union { _Float16 h[2]; unsigned u; } t;
    t.h[0] = (_Float16)lo; t.h[1] = (_Float16)hi;
    return t.u;
}

// Async DMA: copy 16 bytes global -> LDS, tracked by ASYNCcnt (ISA §15.18 op 98).
// vdst VGPR carries the LDS byte offset; vaddr is the 64-bit global address.
__device__ __forceinline__ void async_copy_b128(const void* gsrc, void* ldst) {
    unsigned loff = (unsigned)(size_t)ldst;   // low 32 bits of generic ptr = LDS offset
    asm volatile("global_load_async_to_lds_b128 %0, %1, off"
                 :: "v"(loff), "v"(gsrc) : "memory");
}
__device__ __forceinline__ void wait_async0() {
    asm volatile("s_wait_asynccnt 0x0" ::: "memory");
}

// Fragment layout maps (wave32, v_wmma_f32_16x16x32_f16):
//   A (16x32, MxK): lane = m + 16*((k>>3)&1), e = (k&7) + 8*(k>>4)
//   B (32x16, KxN): lane = (n&15) + 16*((k>>3)&1), e = (k&7) + 8*(k>>4)
//   C/D (16x16):    lane = n + 16*(m>>3),      vgpr i -> m = i + 8*(lane>>4)

// ---------------------------------------------------------------------------
// GEMM: Y = X @ W + b. Block = 128 thr (4 waves), tile 64(M) x 64(N), K step 32.
// Wave w owns M-slice w; 4 accumulators over N; 4 WMMAs per K step.
// HEAD_OUT: f16 output scattered to [B,H,S,Dk]; else fp32 [M,N].
// ---------------------------------------------------------------------------
template <bool HEAD_OUT>
__global__ void __launch_bounds__(128)
gemm64x64(const float* __restrict__ X,    // [BS, D_MODEL]
          const float* __restrict__ W,    // [D_MODEL, D_MODEL]
          const float* __restrict__ bias, // [D_MODEL]
          void* __restrict__ Yv)
{
    __shared__ __align__(32) _Float16 Af[4][32][16];  // [m-slice][lane][e]
    __shared__ __align__(32) _Float16 Bf[4][32][16];  // [n-tile ][lane][e]
    const int tid    = threadIdx.x;
    const int wave   = tid >> 5;
    const int lane   = tid & 31;
    const int laneHi = lane >> 4;
    const int lane16 = lane & 15;
    const int mBase  = blockIdx.x * 64;
    const int nBase  = blockIdx.y * 64;

    v8f acc0 = {}, acc1 = {}, acc2 = {}, acc3 = {};

    for (int k0 = 0; k0 < D_MODEL; k0 += 32) {
        if (k0 + 32 < D_MODEL)
            __builtin_prefetch(X + (size_t)(mBase + (tid >> 1)) * D_MODEL + k0 + 32, 0, 1);

        // A tile 64x32 fp32 -> f16 fragments. 512 float4-groups, 4/thread.
        #pragma unroll
        for (int i = 0; i < 4; ++i) {
            int g  = tid + i * 128;
            int m  = g >> 3;
            int kg = (g & 7) << 2;                    // 0,4,...,28
            const float4 v = *(const float4*)(X + (size_t)(mBase + m) * D_MODEL + k0 + kg);
            int ln = (m & 15) + 16 * ((kg >> 3) & 1);
            int e0 = (kg & 7) + 8 * (kg >> 4);        // multiple of 4
            *(uint2*)(&Af[m >> 4][ln][e0]) = make_uint2(pk2(v.x, v.y), pk2(v.z, v.w));
        }
        // B tile 32x64 fp32 -> f16 fragments, packing rows k,k+1 into dwords.
        #pragma unroll
        for (int i = 0; i < 2; ++i) {
            int t2 = tid + i * 128;                   // 0..255
            int k  = (t2 >> 4) << 1;                  // even 0..30
            int n0 = (t2 & 15) << 2;                  // 0,4,...,60
            const float4 w0 = *(const float4*)(W + (size_t)(k0 + k    ) * D_MODEL + nBase + n0);
            const float4 w1 = *(const float4*)(W + (size_t)(k0 + k + 1) * D_MODEL + nBase + n0);
            int e   = (k & 7) + 8 * (k >> 4);         // even -> dword aligned
            int lhW = 16 * ((k >> 3) & 1);
            int nt  = n0 >> 4;
            *(unsigned*)(&Bf[nt][((n0 + 0) & 15) + lhW][e]) = pk2(w0.x, w1.x);
            *(unsigned*)(&Bf[nt][((n0 + 1) & 15) + lhW][e]) = pk2(w0.y, w1.y);
            *(unsigned*)(&Bf[nt][((n0 + 2) & 15) + lhW][e]) = pk2(w0.z, w1.z);
            *(unsigned*)(&Bf[nt][((n0 + 3) & 15) + lhW][e]) = pk2(w0.w, w1.w);
        }
        __syncthreads();

        const v16h a  = *(const v16h*)(Af[wave][lane]);
        const v16h b0 = *(const v16h*)(Bf[0][lane]);
        const v16h b1 = *(const v16h*)(Bf[1][lane]);
        const v16h b2 = *(const v16h*)(Bf[2][lane]);
        const v16h b3 = *(const v16h*)(Bf[3][lane]);
        acc0 = __builtin_amdgcn_wmma_f32_16x16x32_f16(false, a, false, b0, (short)0, acc0, false, false);
        acc1 = __builtin_amdgcn_wmma_f32_16x16x32_f16(false, a, false, b1, (short)0, acc1, false, false);
        acc2 = __builtin_amdgcn_wmma_f32_16x16x32_f16(false, a, false, b2, (short)0, acc2, false, false);
        acc3 = __builtin_amdgcn_wmma_f32_16x16x32_f16(false, a, false, b3, (short)0, acc3, false, false);
        __syncthreads();
    }

    auto store_tile = [&](const v8f& acc, int nt) {
        const int   col = nBase + nt * 16 + lane16;
        const float bvv = bias[col];
        if constexpr (HEAD_OUT) {
            _Float16* Y = (_Float16*)Yv;
            const int h = col >> 6, d = col & 63;
            #pragma unroll
            for (int i = 0; i < 8; ++i) {
                int row = mBase + wave * 16 + i + 8 * laneHi;
                int bb  = row >> 11;                  // / SEQ
                int s   = row & (SEQ - 1);
                Y[(((size_t)(bb * N_HEADS + h) * SEQ + s) * D_K) + d] = (_Float16)(acc[i] + bvv);
            }
        } else {
            float* Y = (float*)Yv;
            #pragma unroll
            for (int i = 0; i < 8; ++i) {
                int row = mBase + wave * 16 + i + 8 * laneHi;
                Y[(size_t)row * D_MODEL + col] = acc[i] + bvv;
            }
        }
    };
    store_tile(acc0, 0); store_tile(acc1, 1); store_tile(acc2, 2); store_tile(acc3, 3);
}

// ---------------------------------------------------------------------------
// Scores: raw = Q@K^T / 8 + rel_bias, masked. Block = 4 waves, 64q x 64k tile.
// Q/K tiles are pure f16 copies -> async DMA global->LDS into fragment layout.
// Wave w owns q-slice w; reuses its 2 Q fragments over 4 key tiles (8 WMMAs).
// ---------------------------------------------------------------------------
__global__ void __launch_bounds__(128)
scores_kernel(const _Float16* __restrict__ Qh,       // [BH,S,Dk]
              const _Float16* __restrict__ Kh,       // [BH,S,Dk]
              const float*    __restrict__ rel_bias, // [32, N_HEADS]
              const int*      __restrict__ mask,     // [B,1,S,S]
              float*          __restrict__ attn)     // [BH,S,S]
{
    __shared__ __align__(32) _Float16 Qf[4][2][32][16];  // [q-slice][d-chunk][lane][e]
    __shared__ __align__(32) _Float16 Kf[4][2][32][16];  // [k-tile ][d-chunk][lane][e]
    __shared__ float btab[32];                           // rel_bias[:, h]
    const int tid    = threadIdx.x;
    const int wave   = tid >> 5;
    const int lane   = tid & 31;
    const int laneHi = lane >> 4;
    const int lane16 = lane & 15;
    const int qBase  = blockIdx.x * 64;
    const int kBase  = blockIdx.y * 64;
    const int bh     = blockIdx.z;
    const int b      = bh >> 4;
    const int h      = bh & 15;

    const _Float16* Qp = Qh + (size_t)bh * SEQ * D_K;
    const _Float16* Kp = Kh + (size_t)bh * SEQ * D_K;

    if (tid < 32) btab[tid] = rel_bias[tid * N_HEADS + h];

    // Q tile 64x64 f16 -> A-fragment layout (async b128 DMA per 8-half group)
    #pragma unroll
    for (int i = 0; i < 4; ++i) {
        int g  = tid + i * 128;                   // 0..511
        int m  = g >> 3;
        int dg = (g & 7) << 3;                    // 0..56 step 8
        int kk = dg & 31;                         // 0,8,16,24
        int ln = (m & 15) + 16 * ((kk >> 3) & 1);
        int e0 = 8 * (kk >> 4);                   // 0 or 8
        async_copy_b128(Qp + (size_t)(qBase + m) * D_K + dg,
                        &Qf[m >> 4][dg >> 5][ln][e0]);
    }
    // K tile 64x64 f16 -> B-fragment layout (contraction dim = d)
    #pragma unroll
    for (int i = 0; i < 4; ++i) {
        int g  = tid + i * 128;
        int kr = g >> 3;                          // key row 0..63
        int dg = (g & 7) << 3;
        int kk = dg & 31;
        int ln = (kr & 15) + 16 * ((kk >> 3) & 1);
        int e0 = 8 * (kk >> 4);                   // 0 or 8
        async_copy_b128(Kp + (size_t)(kBase + kr) * D_K + dg,
                        &Kf[kr >> 4][dg >> 5][ln][e0]);
    }
    wait_async0();
    __syncthreads();

    const v16h a0 = *(const v16h*)(Qf[wave][0][lane]);
    const v16h a1 = *(const v16h*)(Qf[wave][1][lane]);
    float* outp = attn + (size_t)bh * SEQ * SEQ;

    #pragma unroll
    for (int kt = 0; kt < 4; ++kt) {
        const v16h b0 = *(const v16h*)(Kf[kt][0][lane]);
        const v16h b1 = *(const v16h*)(Kf[kt][1][lane]);
        v8f acc = {};
        acc = __builtin_amdgcn_wmma_f32_16x16x32_f16(false, a0, false, b0, (short)0, acc, false, false);
        acc = __builtin_amdgcn_wmma_f32_16x16x32_f16(false, a1, false, b1, (short)0, acc, false, false);

        const int kpos = kBase + kt * 16 + lane16;
        #pragma unroll
        for (int i = 0; i < 8; ++i) {
            int qpos = qBase + wave * 16 + i + 8 * laneHi;
            int rel  = kpos - qpos;
            rel = rel < -16 ? -16 : (rel > 16 ? 16 : rel);
            rel += 16;                  // [0,32]
            if (rel > 31) rel = 31;     // jnp.take mode="clip"
            float sc = acc[i] * 0.125f + btab[rel];
            int mv = mask[((size_t)b * SEQ + qpos) * SEQ + kpos];
            sc = (mv == 0) ? -1e9f : sc;
            outp[(size_t)qpos * SEQ + kpos] = sc;
        }
    }
}

// ---------------------------------------------------------------------------
// Softmax in place over rows of 2048; one block (256 thr) per row.
// ---------------------------------------------------------------------------
__global__ void __launch_bounds__(256)
softmax_kernel(float* __restrict__ attn)
{
    __shared__ float red[256];
    const int tid = threadIdx.x;
    float* row = attn + (size_t)blockIdx.x * SEQ;

    float v[8];
    #pragma unroll
    for (int i = 0; i < 2; ++i) {
        float4 x = *(const float4*)(row + tid * 8 + i * 4);
        v[i * 4 + 0] = x.x; v[i * 4 + 1] = x.y;
        v[i * 4 + 2] = x.z; v[i * 4 + 3] = x.w;
    }

    float m = -3.4e38f;
    #pragma unroll
    for (int i = 0; i < 8; ++i) m = fmaxf(m, v[i]);
    red[tid] = m; __syncthreads();
    for (int s = 128; s > 0; s >>= 1) {
        if (tid < s) red[tid] = fmaxf(red[tid], red[tid + s]);
        __syncthreads();
    }
    m = red[0];
    __syncthreads();

    float sum = 0.f;
    #pragma unroll
    for (int i = 0; i < 8; ++i) { v[i] = __expf(v[i] - m); sum += v[i]; }
    red[tid] = sum; __syncthreads();
    for (int s = 128; s > 0; s >>= 1) {
        if (tid < s) red[tid] += red[tid + s];
        __syncthreads();
    }
    const float inv = 1.0f / red[0];

    #pragma unroll
    for (int i = 0; i < 2; ++i) {
        float4 x;
        x.x = v[i * 4 + 0] * inv; x.y = v[i * 4 + 1] * inv;
        x.z = v[i * 4 + 2] * inv; x.w = v[i * 4 + 3] * inv;
        *(float4*)(row + tid * 8 + i * 4) = x;
    }
}

// ---------------------------------------------------------------------------
// Context = attn @ V. Block = 4 waves, 64(q) x 64(Dk), K-loop over S step 32.
// ---------------------------------------------------------------------------
__global__ void __launch_bounds__(128)
context_kernel(const float*    __restrict__ attn, // [BH,S,S] probabilities
               const _Float16* __restrict__ Vh,   // [BH,S,Dk]
               float*          __restrict__ ctx)  // [B,S,D_MODEL]
{
    __shared__ __align__(32) _Float16 Pf[4][32][16];  // [q-slice][lane][e]
    __shared__ __align__(32) _Float16 Vf[4][32][16];  // [n-tile ][lane][e]
    const int tid    = threadIdx.x;
    const int wave   = tid >> 5;
    const int lane   = tid & 31;
    const int laneHi = lane >> 4;
    const int lane16 = lane & 15;
    const int qBase  = blockIdx.x * 64;
    const int bh     = blockIdx.y;
    const int b      = bh >> 4;
    const int h      = bh & 15;

    const float*    P  = attn + (size_t)bh * SEQ * SEQ;
    const _Float16* Vp = Vh   + (size_t)bh * SEQ * D_K;

    v8f acc0 = {}, acc1 = {}, acc2 = {}, acc3 = {};

    for (int k0 = 0; k0 < SEQ; k0 += 32) {
        if (k0 + 32 < SEQ)
            __builtin_prefetch(P + (size_t)(qBase + (tid >> 1)) * SEQ + k0 + 32, 0, 1);

        // probs 64x32 fp32 -> f16 A fragments
        #pragma unroll
        for (int i = 0; i < 4; ++i) {
            int g  = tid + i * 128;
            int m  = g >> 3;
            int kg = (g & 7) << 2;
            const float4 v = *(const float4*)(P + (size_t)(qBase + m) * SEQ + k0 + kg);
            int ln = (m & 15) + 16 * ((kg >> 3) & 1);
            int e0 = (kg & 7) + 8 * (kg >> 4);
            *(uint2*)(&Pf[m >> 4][ln][e0]) = make_uint2(pk2(v.x, v.y), pk2(v.z, v.w));
        }
        // V 32x64 f16 -> B fragments: pack rows k,k+1; 128 tasks, 1/thread
        {
            int k  = (tid >> 3) << 1;             // even 0..30
            int n0 = (tid & 7) << 3;              // 0..56 step 8
            const uint4 r0 = *(const uint4*)(Vp + (size_t)(k0 + k    ) * D_K + n0);
            const uint4 r1 = *(const uint4*)(Vp + (size_t)(k0 + k + 1) * D_K + n0);
            const unsigned* u0 = (const unsigned*)&r0;
            const unsigned* u1 = (const unsigned*)&r1;
            int e   = (k & 7) + 8 * (k >> 4);     // even
            int lhW = 16 * ((k >> 3) & 1);
            #pragma unroll
            for (int j = 0; j < 8; ++j) {
                unsigned lo = (u0[j >> 1] >> (16 * (j & 1))) & 0xffffu;
                unsigned hi = (u1[j >> 1] >> (16 * (j & 1))) & 0xffffu;
                int n = n0 + j;
                *(unsigned*)(&Vf[n >> 4][(n & 15) + lhW][e]) = lo | (hi << 16);
            }
        }
        __syncthreads();

        const v16h a  = *(const v16h*)(Pf[wave][lane]);
        const v16h b0 = *(const v16h*)(Vf[0][lane]);
        const v16h b1 = *(const v16h*)(Vf[1][lane]);
        const v16h b2 = *(const v16h*)(Vf[2][lane]);
        const v16h b3 = *(const v16h*)(Vf[3][lane]);
        acc0 = __builtin_amdgcn_wmma_f32_16x16x32_f16(false, a, false, b0, (short)0, acc0, false, false);
        acc1 = __builtin_amdgcn_wmma_f32_16x16x32_f16(false, a, false, b1, (short)0, acc1, false, false);
        acc2 = __builtin_amdgcn_wmma_f32_16x16x32_f16(false, a, false, b2, (short)0, acc2, false, false);
        acc3 = __builtin_amdgcn_wmma_f32_16x16x32_f16(false, a, false, b3, (short)0, acc3, false, false);
        __syncthreads();
    }

    auto st = [&](const v8f& acc, int nt) {
        int dcol = nt * 16 + lane16;
        #pragma unroll
        for (int i = 0; i < 8; ++i) {
            int q = qBase + wave * 16 + i + 8 * laneHi;
            ctx[((size_t)(b * SEQ + q)) * D_MODEL + h * D_K + dcol] = acc[i];
        }
    };
    st(acc0, 0); st(acc1, 1); st(acc2, 2); st(acc3, 3);
}

// ---------------------------------------------------------------------------
// Launch. Inputs: x, mask, Wq, bq, Wk, bk, Wv, bv, Wo, bo, rel_bias.
// d_out = [ out: BS*D_MODEL fp32 | attention: BH*S*S fp32 ].
// Workspace: Qh(8MB) | Kh(8MB) | Vh(8MB) | ctx(16MB) = 40MB.
// ---------------------------------------------------------------------------
extern "C" void kernel_launch(void* const* d_in, const int* in_sizes, int n_in,
                              void* d_out, int out_size, void* d_ws, size_t ws_size,
                              hipStream_t stream) {
    (void)in_sizes; (void)n_in; (void)out_size; (void)ws_size;

    const float* x    = (const float*)d_in[0];
    const int*   mask = (const int*)  d_in[1];
    const float* Wq   = (const float*)d_in[2];
    const float* bq   = (const float*)d_in[3];
    const float* Wk   = (const float*)d_in[4];
    const float* bk   = (const float*)d_in[5];
    const float* Wv   = (const float*)d_in[6];
    const float* bv   = (const float*)d_in[7];
    const float* Wo   = (const float*)d_in[8];
    const float* bo   = (const float*)d_in[9];
    const float* rel  = (const float*)d_in[10];

    float* out  = (float*)d_out;
    float* attn = out + (size_t)BS * D_MODEL;

    char* ws = (char*)d_ws;
    _Float16* Qh  = (_Float16*)(ws);
    _Float16* Kh  = (_Float16*)(ws + (8ull  << 20));
    _Float16* Vh  = (_Float16*)(ws + (16ull << 20));
    float*    ctx = (float*)   (ws + (24ull << 20));

    dim3 gProj(BS / 64, D_MODEL / 64);
    gemm64x64<true><<<gProj, 128, 0, stream>>>(x, Wq, bq, (void*)Qh);
    gemm64x64<true><<<gProj, 128, 0, stream>>>(x, Wk, bk, (void*)Kh);
    gemm64x64<true><<<gProj, 128, 0, stream>>>(x, Wv, bv, (void*)Vh);

    dim3 gS(SEQ / 64, SEQ / 64, BH);
    scores_kernel<<<gS, 128, 0, stream>>>(Qh, Kh, rel, mask, attn);

    softmax_kernel<<<BH * SEQ, 256, 0, stream>>>(attn);

    dim3 gC(SEQ / 64, BH);
    context_kernel<<<gC, 128, 0, stream>>>(attn, Vh, ctx);

    gemm64x64<false><<<gProj, 128, 0, stream>>>(ctx, Wo, bo, (void*)out);
}
